// G_Spatial_50225347559667
// MI455X (gfx1250) — matrix operand order repaired and checked by reading, not compiled, not run on hardware
//
#include <hip/hip_runtime.h>
#include <stdint.h>

// ---------------------------------------------------------------------------
// Problem dims (fixed by the reference)
// ---------------------------------------------------------------------------
static constexpr int Bb = 128, Tt = 25, Vv = 10000, Ee = 512, Hh = 1024, Ll = 196, Aa = 3000;

typedef __bf16 bf16_t;
typedef __attribute__((ext_vector_type(8)))  bf16_t v8bf;
typedef __attribute__((ext_vector_type(16))) bf16_t v16bf;
typedef __attribute__((ext_vector_type(8)))  float  v8f;
typedef unsigned int u32x4 __attribute__((ext_vector_type(4)));
typedef int          i32x4 __attribute__((ext_vector_type(4)));
typedef int          i32x8 __attribute__((ext_vector_type(8)));

__device__ __forceinline__ bf16_t f2bf(float f) {
  union { float f; uint32_t u; } in; in.f = f;
  uint32_t r = (in.u + 0x7FFFu + ((in.u >> 16) & 1u)) >> 16;  // RNE
  union { unsigned short s; bf16_t b; } out; out.s = (unsigned short)r;
  return out.b;
}

// ---------------------------------------------------------------------------
// TDM: load a 32x128 bf16 tile (rows of a [K,N] bf16 matrix starting at gptr)
// into LDS at lds_off, with 16B padding inserted after every 256B row so the
// LDS layout has a 272B (136-element) row stride == BST below.
//   group1: data_size=2B, pad_enable, pad_interval=5 (64 DW), pad_amount=3 (4 DW)
//   tensor dims: dim0=N, dim1=K (for OOB), dim0_stride=N; tile 128 x 32.
// Issued by one wave; completion via s_wait_tensorcnt.
// ---------------------------------------------------------------------------
__device__ __forceinline__ void tdm_load_b_tile(const bf16_t* gptr, unsigned lds_off,
                                                int N, int K)
{
  unsigned long long ga = (unsigned long long)(uintptr_t)gptr;
  u32x4 g0;
  g0[0] = 1u;                                        // count=1, user mode
  g0[1] = lds_off;                                   // lds_addr (bytes)
  g0[2] = (unsigned)(ga & 0xFFFFFFFFu);              // global_addr[31:0]
  g0[3] = (unsigned)((ga >> 32) & 0x01FFFFFFu)       // global_addr[56:32]
        | (2u << 30);                                // type = 2 ("image")
  i32x8 g1;
  g1[0] = (1 << 16)                                  // data_size = 2 bytes
        | (1 << 20)                                  // pad_enable
        | (5 << 22)                                  // pad_interval: 64 DWORDs
        | (3 << 25);                                 // pad_amount: 4 DWORDs (16B)
  g1[1] = (int)(((unsigned)N & 0xFFFFu) << 16);      // tensor_dim0[15:0]
  g1[2] = (int)((((unsigned)N >> 16) & 0xFFFFu)      // tensor_dim0[31:16]
        | (((unsigned)K & 0xFFFFu) << 16));          // tensor_dim1[15:0]
  g1[3] = (int)((((unsigned)K >> 16) & 0xFFFFu)      // tensor_dim1[31:16]
        | (128u << 16));                             // tile_dim0 = 128
  g1[4] = 32;                                        // tile_dim1 = 32
  g1[5] = N;                                         // tensor_dim0_stride[31:0]
  g1[6] = 0;                                         // stride hi / dim1_stride lo
  g1[7] = 0;
  i32x4 z4 = {0, 0, 0, 0};
#if __clang_major__ >= 23
  i32x8 z8 = {0, 0, 0, 0, 0, 0, 0, 0};
  __builtin_amdgcn_tensor_load_to_lds(g0, g1, z4, z4, z8, 0);
#else
  __builtin_amdgcn_tensor_load_to_lds(g0, g1, z4, z4, 0);
#endif
}

// ---------------------------------------------------------------------------
// WMMA GEMM:  Cout[M,N] = act( A[M,K]f32 @ B[K,N]bf16 + bias + Cin )
// A staged f32->bf16 into LDS; B DMA'd by the Tensor Data Mover (fast path)
// or guarded manual loads (edge tiles). Block tile 128x128, K-step 32,
// 8 waves (wave32), wave tile 64x32 = 4x2 WMMA fragments.
// ACT: 0=none, 1=relu, 2=tanh.
// ---------------------------------------------------------------------------
template<int ACT>
__launch_bounds__(256)
__global__ void gemm_wmma_f32(const float* __restrict__ A, const bf16_t* __restrict__ Bm,
                              const float* __restrict__ bias, const float* __restrict__ Cin,
                              float* __restrict__ Cout, int M, int N, int K)
{
  constexpr int BM = 128, BN = 128, BK = 32;
  constexpr int AST = BK + 8;   // 40 bf16 (80B rows, 16B aligned)
  constexpr int BST = BN + 8;   // 136 bf16 (272B rows) == TDM padded stride
  __shared__ __align__(16) bf16_t As[BM * AST];
  __shared__ __align__(16) bf16_t Bs[BK * BST];

  const int tid  = threadIdx.x;
  const int lane = tid & 31;
  const int wave = tid >> 5;
  const int wm   = wave & 1;    // 0..1 -> 64 rows each
  const int wn   = wave >> 1;   // 0..3 -> 32 cols each
  const int row0 = blockIdx.y * BM;
  const int col0 = blockIdx.x * BN;

  v8f acc[4][2];
  #pragma unroll
  for (int i = 0; i < 4; ++i)
    #pragma unroll
    for (int j = 0; j < 2; ++j) {
      v8f z = {0.f,0.f,0.f,0.f,0.f,0.f,0.f,0.f};
      acc[i][j] = z;
    }

  // Per-lane ISA layout constants (16-bit A 16x32; B 32x16)
  const int kb   = (lane < 16) ? 0 : 8;   // A-frag K base for this half-wave
  const int arow = lane & 15;             // A-frag M within tile

  const bool fullN = (col0 + BN) <= N;
  const unsigned bs_lds = (unsigned)(uintptr_t)(void*)&Bs[0];  // LDS byte offset

  for (int k0 = 0; k0 < K; k0 += BK) {
    const bool fullK = (k0 + BK) <= K;
    const bool fastB = fullK && fullN;

    // ---- B tile (32x128 bf16): TDM DMA on fast path, manual on edges ----
    if (fastB) {
      if (wave == 0)
        tdm_load_b_tile(Bm + (size_t)k0 * N + col0, bs_lds, N, K);
    } else {
      const int cg = (tid & 31) * 4;
      #pragma unroll
      for (int r = 0; r < 4; ++r) {
        const int row = r * 8 + (tid >> 5);  // 0..31
        bf16_t* dst = &Bs[row * BST + cg];
        #pragma unroll
        for (int e = 0; e < 4; ++e) {
          bf16_t v = f2bf(0.f);
          if (k0 + row < K && col0 + cg + e < N)
            v = Bm[(size_t)(k0 + row) * N + col0 + cg + e];
          dst[e] = v;
        }
      }
    }

    // ---- stage A tile (128x32) f32 -> bf16 LDS (overlaps the TDM DMA) ----
    {
      const int cg = (tid & 7) * 4;          // col group 0..28
      #pragma unroll
      for (int r = 0; r < 4; ++r) {
        const int row = r * 32 + (tid >> 3); // 0..127
        const float* src = A + (size_t)(row0 + row) * K + k0 + cg;
        bf16_t* dst = &As[row * AST + cg];
        if (fullK) {
          float4 v = *(const float4*)src;
          dst[0] = f2bf(v.x); dst[1] = f2bf(v.y); dst[2] = f2bf(v.z); dst[3] = f2bf(v.w);
        } else {
          #pragma unroll
          for (int e = 0; e < 4; ++e) {
            float v = (k0 + cg + e < K) ? src[e] : 0.f;
            dst[e] = f2bf(v);
          }
        }
      }
    }

    // Speculative prefetch of next K tiles (gfx1250 global_prefetch_b8)
    if (k0 + BK < K) {
      __builtin_prefetch(A  + (size_t)(row0 + (tid >> 3)) * K + k0 + BK, 0, 1);
      __builtin_prefetch(Bm + (size_t)(k0 + BK + (tid >> 5)) * N + col0 + (tid & 31) * 4, 0, 1);
    }

    // TDM completion (issuing wave), then release the workgroup
    if (fastB && wave == 0)
      __builtin_amdgcn_s_wait_tensorcnt(0);
    __syncthreads();

    // ---- load fragments from LDS (matches ISA per-lane VGPR layout) ----
    v16bf afr[4];
    #pragma unroll
    for (int mt = 0; mt < 4; ++mt) {
      const bf16_t* p = &As[(wm * 64 + mt * 16 + arow) * AST + kb];
      v8bf lo = *(const v8bf*)p;           // K = kb .. kb+7
      v8bf hi = *(const v8bf*)(p + 16);    // K = kb+16 .. kb+23
      v16bf a;
      #pragma unroll
      for (int e = 0; e < 8; ++e) { a[e] = lo[e]; a[e + 8] = hi[e]; }
      afr[mt] = a;
    }
    v16bf bfr[2];
    #pragma unroll
    for (int nt = 0; nt < 2; ++nt) {
      const bf16_t* p = &Bs[kb * BST + wn * 32 + nt * 16 + arow];  // col N = lane&15, K = kb..kb+7, kb+16..kb+23
      v16bf b;
      #pragma unroll
      for (int e = 0; e < 8; ++e) { b[e] = p[e * BST]; b[e + 8] = p[(e + 16) * BST]; }
      bfr[nt] = b;
    }

    // ---- 8 WMMAs per K-step ----
    #pragma unroll
    for (int mt = 0; mt < 4; ++mt)
      #pragma unroll
      for (int nt = 0; nt < 2; ++nt)
        acc[mt][nt] = __builtin_amdgcn_wmma_f32_16x16x32_bf16(
            false, afr[mt], false, bfr[nt], (short)0, acc[mt][nt], false, false);

    __syncthreads();
  }

  // ---- epilogue: C/D layout (VGPR r -> M=r(+8 hi half-wave), N=lane&15) ----
  const int mofs = (lane >= 16) ? 8 : 0;
  const int ncol = lane & 15;
  #pragma unroll
  for (int mt = 0; mt < 4; ++mt) {
    #pragma unroll
    for (int nt = 0; nt < 2; ++nt) {
      #pragma unroll
      for (int r = 0; r < 8; ++r) {
        const int grow = row0 + wm * 64 + mt * 16 + mofs + r;
        const int gcol = col0 + wn * 32 + nt * 16 + ncol;
        if (grow < M && gcol < N) {
          float v = acc[mt][nt][r];
          if (Cin)  v += Cin[(size_t)grow * N + gcol];
          if (bias) v += bias[gcol];
          if (ACT == 1) v = fmaxf(v, 0.f);
          if (ACT == 2) v = tanhf(v);
          Cout[(size_t)grow * N + gcol] = v;
        }
      }
    }
  }
}

// ---------------------------------------------------------------------------
// Spatial attention: one block per batch element.
// ---------------------------------------------------------------------------
__launch_bounds__(256)
__global__ void attention_kernel(const float* __restrict__ feats,  // [B,L,H]
                                 const float* __restrict__ q,      // [B,H]
                                 float* __restrict__ ctx)          // [B,H]
{
  const int b = blockIdx.x;
  const int tid = threadIdx.x;
  const int lane = tid & 31;
  const int wave = tid >> 5;
  __shared__ float sc[Ll];
  __shared__ float red[256];
  const float* fb = feats + (size_t)b * Ll * Hh;
  const float* qb = q + (size_t)b * Hh;

  for (int l = wave; l < Ll; l += 8) {
    float p = 0.f;
    const float* fr = fb + (size_t)l * Hh;
    for (int h = lane; h < Hh; h += 32) p += fr[h] * qb[h];
    #pragma unroll
    for (int o = 16; o > 0; o >>= 1) p += __shfl_xor(p, o, 32);
    if (lane == 0) sc[l] = p;
  }
  __syncthreads();

  float m = -3.4e38f;
  for (int i = tid; i < Ll; i += 256) m = fmaxf(m, sc[i]);
  red[tid] = m; __syncthreads();
  for (int s = 128; s > 0; s >>= 1) {
    if (tid < s) red[tid] = fmaxf(red[tid], red[tid + s]);
    __syncthreads();
  }
  m = red[0]; __syncthreads();

  float ssum = 0.f;
  for (int i = tid; i < Ll; i += 256) { float e = __expf(sc[i] - m); sc[i] = e; ssum += e; }
  red[tid] = ssum; __syncthreads();
  for (int s = 128; s > 0; s >>= 1) {
    if (tid < s) red[tid] += red[tid + s];
    __syncthreads();
  }
  const float inv = 1.f / red[0];
  __syncthreads();
  for (int i = tid; i < Ll; i += 256) sc[i] *= inv;
  __syncthreads();

  for (int h = tid; h < Hh; h += 256) {
    float a = 0.f;
    for (int l = 0; l < Ll; ++l) a += sc[l] * fb[(size_t)l * Hh + h];
    ctx[(size_t)b * Hh + h] = a;
  }
}

// ---------------------------------------------------------------------------
// LSTM pointwise + small helpers
// ---------------------------------------------------------------------------
__global__ void lstm_pointwise(const float* __restrict__ pre,  // [B,4H]
                               const float* __restrict__ cin,  // [B,H]
                               float* __restrict__ h, float* __restrict__ c)
{
  const int idx = blockIdx.x * blockDim.x + threadIdx.x;
  if (idx >= Bb * Hh) return;
  const int b = idx / Hh, j = idx - b * Hh;
  const float* p = pre + (size_t)b * 4 * Hh;
  const float gi = p[j], gf = p[Hh + j], gg = p[2 * Hh + j], go = p[3 * Hh + j];
  const float si = 1.f / (1.f + __expf(-gi));
  const float sf = 1.f / (1.f + __expf(-gf));
  const float so = 1.f / (1.f + __expf(-go));
  const float cy = sf * cin[idx] + si * tanhf(gg);
  c[idx] = cy;
  h[idx] = so * tanhf(cy);
}

__global__ void combine_store(const float* __restrict__ h2, const float* __restrict__ c2,
                              const float* __restrict__ h1, const float* __restrict__ c1,
                              float* __restrict__ res_t)
{
  const int idx = blockIdx.x * blockDim.x + threadIdx.x;
  if (idx >= Bb * 2 * Hh) return;
  const int b = idx / (2 * Hh), j = idx - b * (2 * Hh);
  float v;
  if (j < Hh) v = h2[b * Hh + j] + c2[b * Hh + j];
  else        v = h1[b * Hh + j - Hh] + c1[b * Hh + j - Hh];
  res_t[idx] = v;
}

__global__ void gather_combined(const float* __restrict__ res, const int* __restrict__ lengths,
                                float* __restrict__ comb)
{
  const int idx = blockIdx.x * blockDim.x + threadIdx.x;
  if (idx >= Bb * 2 * Hh) return;
  const int b = idx / (2 * Hh);
  int L = lengths[b]; L = (L < 1) ? 1 : ((L > Tt) ? Tt : L);
  comb[idx] = res[(size_t)(L - 1) * Bb * 2 * Hh + idx];
}

__global__ void permute_emb(const float* __restrict__ src, float* __restrict__ embA)
{
  const int idx = blockIdx.x * blockDim.x + threadIdx.x;
  if (idx >= Bb * Tt * Ee) return;
  const int e = idx % Ee;
  const int bt = idx / Ee;
  const int t = bt % Tt, b = bt / Tt;
  embA[((size_t)(1 + t) * Bb + b) * Ee + e] = src[idx];
}

__global__ void copy_f32(const float* __restrict__ s, float* __restrict__ d, int n)
{
  const int i = blockIdx.x * blockDim.x + threadIdx.x;
  if (i < n) d[i] = s[i];
}

__global__ void cvt_f32_bf16(const float* __restrict__ s, bf16_t* __restrict__ d, int n)
{
  const int i = blockIdx.x * blockDim.x + threadIdx.x;
  if (i < n) d[i] = f2bf(s[i]);
}

// ---------------------------------------------------------------------------
// Orchestration
// ---------------------------------------------------------------------------
extern "C" void kernel_launch(void* const* d_in, const int* in_sizes, int n_in,
                              void* d_out, int out_size, void* d_ws, size_t ws_size,
                              hipStream_t stream)
{
  const float* feat_g   = (const float*)d_in[0];   // [B,E]
  const float* feat_l   = (const float*)d_in[1];   // [B,L,H]
  const float* captions = (const float*)d_in[2];   // [B,T,V]
  const int*   lengths  = (const int*)  d_in[3];   // [B]
  const float* h0       = (const float*)d_in[4];
  const float* c0       = (const float*)d_in[5];
  const float* embW     = (const float*)d_in[6];   // [V,E]
  const float* embB     = (const float*)d_in[7];   // [E]
  struct Cell { const float *Wi, *Wh, *Wc, *b, *Wa, *ba; };
  const Cell cl1 = {(const float*)d_in[8],  (const float*)d_in[9],  (const float*)d_in[10],
                    (const float*)d_in[11], (const float*)d_in[12], (const float*)d_in[13]};
  const Cell cl2 = {(const float*)d_in[14], (const float*)d_in[15], (const float*)d_in[16],
                    (const float*)d_in[17], (const float*)d_in[18], (const float*)d_in[19]};
  const float* fc1W = (const float*)d_in[20]; const float* fc1b = (const float*)d_in[21];
  const float* fc2W = (const float*)d_in[22]; const float* fc2b = (const float*)d_in[23];
  const float* fc3W = (const float*)d_in[24]; const float* fc3b = (const float*)d_in[25];

  // ---- workspace carve-out ----
  char* ws = (char*)d_ws;
  size_t off = 0;
  auto alloc = [&](size_t bytes) -> void* {
    void* p = ws + off;
    off += (bytes + 255) & ~(size_t)255;
    return p;
  };
  // bf16 weight copies (converted once per launch; streamed 25x from L2 after)
  bf16_t* embW_bf = (bf16_t*)alloc((size_t)Vv * Ee * 2);
  bf16_t* Wi1_bf  = (bf16_t*)alloc((size_t)Ee * 4 * Hh * 2);
  bf16_t* Wh1_bf  = (bf16_t*)alloc((size_t)Hh * 4 * Hh * 2);
  bf16_t* Wc1_bf  = (bf16_t*)alloc((size_t)Hh * 4 * Hh * 2);
  bf16_t* Wa1_bf  = (bf16_t*)alloc((size_t)Hh * Hh * 2);
  bf16_t* Wi2_bf  = (bf16_t*)alloc((size_t)Ee * 4 * Hh * 2);
  bf16_t* Wh2_bf  = (bf16_t*)alloc((size_t)Hh * 4 * Hh * 2);
  bf16_t* Wc2_bf  = (bf16_t*)alloc((size_t)Hh * 4 * Hh * 2);
  bf16_t* Wa2_bf  = (bf16_t*)alloc((size_t)Hh * Hh * 2);
  bf16_t* fc1W_bf = (bf16_t*)alloc((size_t)2 * Hh * 2 * Hh * 2);
  bf16_t* fc2W_bf = (bf16_t*)alloc((size_t)2 * Hh * 2 * Hh * 2);
  bf16_t* fc3W_bf = (bf16_t*)alloc((size_t)2 * Hh * Aa * 2);
  // f32 activations / intermediates
  float* embTmp = (float*)alloc((size_t)Bb * Tt * Ee * 4);         // [B*T,E]
  float* embA   = (float*)alloc((size_t)(Tt + 1) * Bb * Ee * 4);   // [T+1,B,E]
  float* Xi1    = (float*)alloc((size_t)Tt * Bb * 4 * Hh * 4);     // [T,B,4H]
  float* Xi2    = (float*)alloc((size_t)Tt * Bb * 4 * Hh * 4);
  float* qbuf   = (float*)alloc((size_t)Bb * Hh * 4);
  float* ctxbuf = (float*)alloc((size_t)Bb * Hh * 4);
  float* prebuf = (float*)alloc((size_t)Bb * 4 * Hh * 4);
  float* hS[2]  = {(float*)alloc((size_t)Bb * Hh * 4), (float*)alloc((size_t)Bb * Hh * 4)};
  float* cS[2]  = {(float*)alloc((size_t)Bb * Hh * 4), (float*)alloc((size_t)Bb * Hh * 4)};
  float* h1b    = (float*)alloc((size_t)Bb * Hh * 4);
  float* c1b    = (float*)alloc((size_t)Bb * Hh * 4);
  float* res    = (float*)alloc((size_t)Tt * Bb * 2 * Hh * 4);     // [T,B,2H]
  float* comb   = (float*)alloc((size_t)Bb * 2 * Hh * 4);
  float* fcb1   = (float*)alloc((size_t)Bb * 2 * Hh * 4);
  float* fcb2   = (float*)alloc((size_t)Bb * 2 * Hh * 4);
  (void)ws_size; (void)in_sizes; (void)n_in; (void)out_size;

  auto cvt = [&](const float* s, bf16_t* d, size_t n) {
    cvt_f32_bf16<<<((int)n + 255) / 256, 256, 0, stream>>>(s, d, (int)n);
  };
  auto gemm = [&](const float* A, const bf16_t* Bm, const float* bias, const float* Cin,
                  float* Cout, int M, int N, int K, int act) {
    dim3 grid((N + 127) / 128, (M + 127) / 128);
    if (act == 0)      gemm_wmma_f32<0><<<grid, 256, 0, stream>>>(A, Bm, bias, Cin, Cout, M, N, K);
    else if (act == 1) gemm_wmma_f32<1><<<grid, 256, 0, stream>>>(A, Bm, bias, Cin, Cout, M, N, K);
    else               gemm_wmma_f32<2><<<grid, 256, 0, stream>>>(A, Bm, bias, Cin, Cout, M, N, K);
  };

  // ---- weight conversions (once per launch) ----
  cvt(embW,   embW_bf, (size_t)Vv * Ee);
  cvt(cl1.Wi, Wi1_bf,  (size_t)Ee * 4 * Hh);
  cvt(cl1.Wh, Wh1_bf,  (size_t)Hh * 4 * Hh);
  cvt(cl1.Wc, Wc1_bf,  (size_t)Hh * 4 * Hh);
  cvt(cl1.Wa, Wa1_bf,  (size_t)Hh * Hh);
  cvt(cl2.Wi, Wi2_bf,  (size_t)Ee * 4 * Hh);
  cvt(cl2.Wh, Wh2_bf,  (size_t)Hh * 4 * Hh);
  cvt(cl2.Wc, Wc2_bf,  (size_t)Hh * 4 * Hh);
  cvt(cl2.Wa, Wa2_bf,  (size_t)Hh * Hh);
  cvt(fc1W,   fc1W_bf, (size_t)2 * Hh * 2 * Hh);
  cvt(fc2W,   fc2W_bf, (size_t)2 * Hh * 2 * Hh);
  cvt(fc3W,   fc3W_bf, (size_t)2 * Hh * Aa);

  int n;
  // embA[0] = features_global; embA[1..T] = relu(captions @ embW + embB)
  n = Bb * Ee;
  copy_f32<<<(n + 255) / 256, 256, 0, stream>>>(feat_g, embA, n);
  gemm(captions, embW_bf, embB, nullptr, embTmp, Bb * Tt, Ee, Vv, /*relu*/1);
  n = Bb * Tt * Ee;
  permute_emb<<<(n + 255) / 256, 256, 0, stream>>>(embTmp, embA);

  // Hoisted input projections for all steps: Xi = x @ Wi + b
  gemm(embA,                   Wi1_bf, cl1.b, nullptr, Xi1, Tt * Bb, 4 * Hh, Ee, 0);
  gemm(embA + (size_t)Bb * Ee, Wi2_bf, cl2.b, nullptr, Xi2, Tt * Bb, 4 * Hh, Ee, 0);

  // init recurrent state
  n = Bb * Hh;
  copy_f32<<<(n + 255) / 256, 256, 0, stream>>>(h0, hS[0], n);
  copy_f32<<<(n + 255) / 256, 256, 0, stream>>>(c0, cS[0], n);

  const int nBH  = Bb * Hh;
  const int nB2H = Bb * 2 * Hh;
  int cur = 0;
  for (int t = 0; t < Tt; ++t) {
    // ---- cell 1 ----
    gemm(hS[cur], Wa1_bf, cl1.ba, nullptr, qbuf, Bb, Hh, Hh, 0);
    attention_kernel<<<Bb, 256, 0, stream>>>(feat_l, qbuf, ctxbuf);
    gemm(hS[cur], Wh1_bf, nullptr, Xi1 + (size_t)t * Bb * 4 * Hh, prebuf, Bb, 4 * Hh, Hh, 0);
    gemm(ctxbuf,  Wc1_bf, nullptr, prebuf,                        prebuf, Bb, 4 * Hh, Hh, 0);
    lstm_pointwise<<<(nBH + 255) / 256, 256, 0, stream>>>(prebuf, cS[cur], h1b, c1b);

    // ---- cell 2 (input = reversed embedding -> Xi2 slice T-1-t, hidden = h1) ----
    gemm(h1b, Wa2_bf, cl2.ba, nullptr, qbuf, Bb, Hh, Hh, 0);
    attention_kernel<<<Bb, 256, 0, stream>>>(feat_l, qbuf, ctxbuf);
    gemm(h1b,    Wh2_bf, nullptr, Xi2 + (size_t)(Tt - 1 - t) * Bb * 4 * Hh, prebuf, Bb, 4 * Hh, Hh, 0);
    gemm(ctxbuf, Wc2_bf, nullptr, prebuf,                                   prebuf, Bb, 4 * Hh, Hh, 0);
    const int nxt = cur ^ 1;
    lstm_pointwise<<<(nBH + 255) / 256, 256, 0, stream>>>(prebuf, c1b, hS[nxt], cS[nxt]);

    combine_store<<<(nB2H + 255) / 256, 256, 0, stream>>>(
        hS[nxt], cS[nxt], h1b, c1b, res + (size_t)t * Bb * 2 * Hh);
    cur = nxt;
  }

  // gather at lengths-1, then FC head
  gather_combined<<<(nB2H + 255) / 256, 256, 0, stream>>>(res, lengths, comb);
  gemm(comb, fc1W_bf, fc1b, nullptr, fcb1, Bb, 2 * Hh, 2 * Hh, /*tanh*/2);
  gemm(fcb1, fc2W_bf, fc2b, nullptr, fcb2, Bb, 2 * Hh, 2 * Hh, /*tanh*/2);
  gemm(fcb2, fc3W_bf, fc3b, nullptr, (float*)d_out, Bb, Aa, 2 * Hh, 0);
}
